// RMABlock_84997402788260
// MI455X (gfx1250) — matrix operand order repaired and checked by reading, not compiled, not run on hardware
//
#include <hip/hip_runtime.h>
#include <hip/hip_bf16.h>

// ---------------- problem constants ----------------
#define BB 4
#define LCC 2048
#define MMEM 256
#define DD 1024
#define HH 16
#define HD 64
#define TT (MMEM + LCC)          // 2304
#define NROWS (BB * TT)          // 9216
#define EPS 1e-5f
#define SP 2312                  // padded LDS row stride (halves) for T=2304
#define SBP 40                   // padded LDS B-row stride (halves) in GEMM staging

typedef _Float16 v8h  __attribute__((ext_vector_type(8)));
typedef _Float16 v16h __attribute__((ext_vector_type(16)));
typedef float    v8f  __attribute__((ext_vector_type(8)));
typedef int      i4v  __attribute__((vector_size(16)));
typedef __attribute__((address_space(1))) i4v* gptr_i4;
typedef __attribute__((address_space(3))) i4v* lptr_i4;

#if __has_builtin(__builtin_amdgcn_global_load_async_to_lds_b128) && \
    __has_builtin(__builtin_amdgcn_s_wait_asynccnt)
#define USE_ASYNC_LDS 1
#else
#define USE_ASYNC_LDS 0
#endif

// Build a 16-half A/B fragment from two contiguous 8-half (16B) chunks.
__device__ __forceinline__ v16h make_frag(const _Float16* p0, const _Float16* p1) {
  v8h a = *(const v8h*)p0;
  v8h b = *(const v8h*)p1;
  return __builtin_shufflevector(a, b, 0,1,2,3,4,5,6,7,8,9,10,11,12,13,14,15);
}
// A-matrix 16x32 f16: lane<16 -> row=lane, k in [k0,k0+16) contiguous
//                     lane>=16 -> row=lane-16, k in [k0+16,k0+32) contiguous
__device__ __forceinline__ v16h load_a(const _Float16* rowbase /* row ptr + k0 */, int hi) {
  return make_frag(rowbase + hi * 16, rowbase + hi * 16 + 8);
}
// B-matrix 32x16 f16: lane holds column (lane&15), k in [k0+hi*16, k0+hi*16+16) contiguous
__device__ __forceinline__ v16h load_b(const _Float16* colbase /* col ptr + k0 + hi*16 */) {
  return make_frag(colbase, colbase + 8);
}

__device__ __forceinline__ v8f wmma_f16(v16h a, v16h b, v8f c) {
  return __builtin_amdgcn_wmma_f32_16x16x32_f16(false, a, false, b, (short)0, c, false, false);
}

// Stage 16 bytes (8 halves) global -> LDS, async when available.
__device__ __forceinline__ void stage16(const _Float16* gsrc, _Float16* lds_dst) {
#if USE_ASYNC_LDS
  __builtin_amdgcn_global_load_async_to_lds_b128(
      (gptr_i4)(void*)gsrc, (lptr_i4)(void*)lds_dst, 0, 0);
#else
  *(v8h*)lds_dst = *(const v8h*)gsrc;
#endif
}

// ---------------- f32 -> f16 conversion ----------------
__global__ __launch_bounds__(256) void cvt_f16_kernel(const float* __restrict__ s,
                                                      _Float16* __restrict__ d, int n) {
  for (int i = blockIdx.x * blockDim.x + threadIdx.x; i < n; i += gridDim.x * blockDim.x)
    d[i] = (_Float16)s[i];
}

// ---------------- LayerNorm helpers ----------------
// One block (256 threads) per token row of length D=1024; each thread owns 4 elems.
__device__ __forceinline__ void ln_row(float x[4], int idx0,
                                       const float* __restrict__ g,
                                       const float* __restrict__ be,
                                       _Float16* __restrict__ outrow) {
  const int lane = threadIdx.x & 31, wid = threadIdx.x >> 5;
  __shared__ float red1[8], red2[8];
  float s = x[0] + x[1] + x[2] + x[3];
#pragma unroll
  for (int off = 16; off; off >>= 1) s += __shfl_xor(s, off, 32);
  if (lane == 0) red1[wid] = s;
  __syncthreads();
  float tot = 0.f;
#pragma unroll
  for (int w = 0; w < 8; ++w) tot += red1[w];
  float mean = tot * (1.f / (float)DD);
  float vs = 0.f;
#pragma unroll
  for (int i = 0; i < 4; ++i) { float dv = x[i] - mean; vs += dv * dv; }
#pragma unroll
  for (int off = 16; off; off >>= 1) vs += __shfl_xor(vs, off, 32);
  if (lane == 0) red2[wid] = vs;
  __syncthreads();
  float totv = 0.f;
#pragma unroll
  for (int w = 0; w < 8; ++w) totv += red2[w];
  float rstd = rsqrtf(totv * (1.f / (float)DD) + EPS);
#pragma unroll
  for (int i = 0; i < 4; ++i) {
    int idx = idx0 + i;
    outrow[idx] = (_Float16)((x[i] - mean) * rstd * g[idx] + be[idx]);
  }
}

// z = concat(mem, chunk) + pos ; zn = LN(z)*g+b   (grid = B*T)
__global__ __launch_bounds__(256)
void build_z_ln_kernel(const float* __restrict__ chunk, const float* __restrict__ mem,
                       const float* __restrict__ pos, const float* __restrict__ g,
                       const float* __restrict__ be, float* __restrict__ z,
                       _Float16* __restrict__ zn) {
  const int row = blockIdx.x;
  const int b = row / TT, t = row % TT;
  const float* src = (t < MMEM) ? (mem + ((size_t)b * MMEM + t) * DD)
                                : (chunk + ((size_t)b * LCC + (t - MMEM)) * DD);
  const float* pp = pos + (size_t)t * DD;
  float* zr = z + (size_t)row * DD;
  const int base = threadIdx.x * 4;
  float x[4];
#pragma unroll
  for (int i = 0; i < 4; ++i) { x[i] = src[base + i] + pp[base + i]; zr[base + i] = x[i]; }
  ln_row(x, base, g, be, zn + (size_t)row * DD);
}

// hn = LN(z)*g+b   (grid = B*T)
__global__ __launch_bounds__(256)
void ln_kernel(const float* __restrict__ z, const float* __restrict__ g,
               const float* __restrict__ be, _Float16* __restrict__ hn) {
  const int row = blockIdx.x;
  const float* zr = z + (size_t)row * DD;
  const int base = threadIdx.x * 4;
  float x[4];
#pragma unroll
  for (int i = 0; i < 4; ++i) x[i] = zr[base + i];
  ln_row(x, base, g, be, hn + (size_t)row * DD);
}

// ---------------- WMMA GEMM with LDS-staged B: out = A[M,K] @ W[N,K]^T + bias --------
// Workgroup (8 waves) computes a 128(M) x 64(N) tile; the 64x32 f16 B chunk is staged
// in LDS once per k-step (async global->LDS, double buffered) and shared by all waves.
// mode 0: f16 store to o0[M,N]
// mode 1: QKV split: q=o0 (scaled 1/8), k=o1 (both [B,T,D]), v -> o2 transposed [B,D,T]
// mode 2: f32 residual accumulate: ofz[M,N] += result
// mode 3: ReLU f16 store to o0[M,N]
__global__ __launch_bounds__(256)
void gemm_bstage_kernel(const _Float16* __restrict__ A, const _Float16* __restrict__ W,
                        const float* __restrict__ bias,
                        _Float16* __restrict__ o0, _Float16* __restrict__ o1,
                        _Float16* __restrict__ o2, float* __restrict__ ofz,
                        int Mrows, int N, int K, int mode) {
  __shared__ _Float16 bsm[2][64 * SBP];          // 2 x 5120 B double buffer
  const int tid  = threadIdx.x;
  const int lane = tid & 31;
  const int wid  = tid >> 5;
  const int hi   = lane >> 4;
  const int ln   = lane & 15;
  const int n64  = N >> 6;
  const int mt   = blockIdx.x / n64;             // 128-row tile index
  if (mt >= (Mrows >> 7)) return;                // block-uniform guard
  const int nt0  = (blockIdx.x % n64) << 6;
  const int mrow = mt * 128 + wid * 16;          // this wave's 16 rows

  // staging assignment: thread copies 8 halves of B chunk (64 rows x 32 k)
  const int sn = tid >> 2;                       // 0..63 (row within N tile)
  const int sk = (tid & 3) * 8;                  // 0,8,16,24 (k offset)
  const _Float16* wsrc = W + (size_t)(nt0 + sn) * K + sk;
  _Float16* sdst0 = &bsm[0][sn * SBP + sk];
  _Float16* sdst1 = &bsm[1][sn * SBP + sk];

  const _Float16* arow = A + (size_t)(mrow + ln) * K;

  v8f acc0 = {}, acc1 = {}, acc2 = {}, acc3 = {};

  stage16(wsrc, sdst0);                          // prologue: k0 = 0 -> buf 0
  int buf = 0;
  for (int k0 = 0; k0 < K; k0 += 32) {
    if (k0 + 32 < K) {
      stage16(wsrc + k0 + 32, buf ? sdst0 : sdst1);   // prefetch next chunk
#if USE_ASYNC_LDS
      __builtin_amdgcn_s_wait_asynccnt(1);            // current chunk landed
#endif
    } else {
#if USE_ASYNC_LDS
      __builtin_amdgcn_s_wait_asynccnt(0);
#endif
    }
    __syncthreads();                                  // chunk visible to all waves

    __builtin_prefetch(arow + k0 + 128, 0, 1);        // global_prefetch_b8 on A stream
    v16h a = load_a(arow + k0, hi);
    const _Float16* bb = &bsm[buf][ln * SBP + hi * 16];
    v16h b0 = load_b(bb);
    v16h b1 = load_b(bb + 16 * SBP);
    v16h b2 = load_b(bb + 32 * SBP);
    v16h b3 = load_b(bb + 48 * SBP);
    acc0 = wmma_f16(a, b0, acc0);
    acc1 = wmma_f16(a, b1, acc1);
    acc2 = wmma_f16(a, b2, acc2);
    acc3 = wmma_f16(a, b3, acc3);

    __syncthreads();                                  // reads done before next overwrite
    buf ^= 1;
  }

  v8f accs[4] = {acc0, acc1, acc2, acc3};
#pragma unroll
  for (int j = 0; j < 4; ++j) {
    const int n = nt0 + j * 16 + ln;
    const float bv = bias[n];
#pragma unroll
    for (int i = 0; i < 8; ++i) {
      const int r = mrow + i + hi * 8;        // C/D layout: vgpr i -> M=i (+8 for hi lanes)
      float val = accs[j][i] + bv;
      if (mode == 0) {
        o0[(size_t)r * N + n] = (_Float16)val;
      } else if (mode == 1) {
        const int which = n >> 10, d = n & 1023;
        if (which == 0)      o0[(size_t)r * DD + d] = (_Float16)(val * 0.125f);
        else if (which == 1) o1[(size_t)r * DD + d] = (_Float16)val;
        else {
          const int b = r / TT, t = r % TT;
          o2[((size_t)b * DD + d) * TT + t] = (_Float16)val;   // V transposed
        }
      } else if (mode == 2) {
        ofz[(size_t)r * N + n] += val;
      } else {
        o0[(size_t)r * N + n] = (_Float16)fmaxf(val, 0.f);
      }
    }
  }
}

// ---------------- attention: one block per (b, h, 16-query tile) ----------------
__global__ __launch_bounds__(256)
void attn_kernel(const _Float16* __restrict__ q, const _Float16* __restrict__ k,
                 const _Float16* __restrict__ vT, _Float16* __restrict__ ao) {
  extern __shared__ _Float16 sm[];                 // [16][SP] logits then probs (f16)
  float* redbuf = (float*)(sm + 16 * SP);          // 4 * 256 floats
  __shared__ float rinv[16];
  const int lane = threadIdx.x & 31, wid = threadIdx.x >> 5;
  const int hi = lane >> 4, ln = lane & 15;
  const int QT = TT / 16;                          // 144
  const int qt = blockIdx.x % QT;
  const int h  = (blockIdx.x / QT) % HH;
  const int b  = blockIdx.x / (QT * HH);

  // phase 1: S = Q K^T (q pre-scaled), mask, stash f16 logits in LDS
  const _Float16* qbase = q + ((size_t)(b * TT + qt * 16 + ln)) * DD + h * HD;
  for (int ct = wid; ct < QT; ct += 8) {
    v8f acc = {};
    const _Float16* kbase = k + ((size_t)(b * TT + ct * 16 + ln)) * DD + h * HD + hi * 16;
#pragma unroll
    for (int kk = 0; kk < HD; kk += 32) {
      v16h a  = load_a(qbase + kk, hi);
      v16h bf = load_b(kbase + kk);
      acc = wmma_f16(a, bf, acc);
    }
    const int kg = ct * 16 + ln;
#pragma unroll
    for (int i = 0; i < 8; ++i) {
      const int m = i + hi * 8;
      const int qg = qt * 16 + m;
      float v = acc[i];
      if (qg >= MMEM && kg >= MMEM && kg > qg) v = -INFINITY;  // reference mask
      sm[m * SP + kg] = (_Float16)v;
    }
  }
  __syncthreads();

  // phase 2: row softmax (unnormalized probs in LDS, 1/sum in rinv)
#pragma unroll
  for (int rr = 0; rr < 2; ++rr) {
    const int r = wid + rr * 8;
    float mx = -INFINITY;
    for (int j = 0; j < TT / 32; ++j)
      mx = fmaxf(mx, (float)sm[r * SP + lane + j * 32]);
#pragma unroll
    for (int off = 16; off; off >>= 1) mx = fmaxf(mx, __shfl_xor(mx, off, 32));
    float sum = 0.f;
    for (int j = 0; j < TT / 32; ++j) {
      const int idx = r * SP + lane + j * 32;
      float p = __expf((float)sm[idx] - mx);
      sum += p;
      sm[idx] = (_Float16)p;
    }
#pragma unroll
    for (int off = 16; off; off >>= 1) sum += __shfl_xor(sum, off, 32);
    if (lane == 0) rinv[r] = 1.0f / sum;
  }
  __syncthreads();

  // phase 3: O = P @ V ; wave w -> hd tile (w&3), key-chunk slice (w>>2)
  const int nt = wid & 3;
  v8f acc = {};
  const _Float16* prow  = sm + ln * SP;
  const _Float16* vbase = vT + ((size_t)b * DD + h * HD + nt * 16 + ln) * TT + hi * 16;
  for (int c = (wid >> 2); c < TT / 32; c += 2) {
    v16h a  = load_a(prow + c * 32, hi);
    v16h bf = load_b(vbase + c * 32);
    acc = wmma_f16(a, bf, acc);
  }
  if (wid >= 4) {
#pragma unroll
    for (int i = 0; i < 8; ++i) redbuf[(wid - 4) * 256 + i * 32 + lane] = acc[i];
  }
  __syncthreads();
  if (wid < 4) {
#pragma unroll
    for (int i = 0; i < 8; ++i) acc[i] += redbuf[wid * 256 + i * 32 + lane];
#pragma unroll
    for (int i = 0; i < 8; ++i) {
      const int m = i + hi * 8;
      const int t = qt * 16 + m;
      const int d = h * HD + nt * 16 + ln;
      ao[((size_t)(b * TT + t)) * DD + d] = (_Float16)(acc[i] * rinv[m]);
    }
  }
}

// ---------------- scatter z -> (h_k, mem_candidate) ----------------
__global__ __launch_bounds__(256)
void write_out_kernel(const float* __restrict__ z, float* __restrict__ out) {
  const int n = BB * TT * DD;
  for (int i = blockIdx.x * blockDim.x + threadIdx.x; i < n; i += gridDim.x * blockDim.x) {
    const int d = i % DD;
    const int t = (i / DD) % TT;
    const int b = i / (DD * TT);
    const float v = z[i];
    if (t < MMEM)
      out[(size_t)BB * LCC * DD + ((size_t)b * MMEM + t) * DD + d] = v;   // mem_candidate
    else
      out[((size_t)b * LCC + (t - MMEM)) * DD + d] = v;                   // h_k
  }
}

// ---------------- host launcher ----------------
extern "C" void kernel_launch(void* const* d_in, const int* in_sizes, int n_in,
                              void* d_out, int out_size, void* d_ws, size_t ws_size,
                              hipStream_t stream) {
  const float* chunk  = (const float*)d_in[0];
  const float* mem    = (const float*)d_in[1];
  const float* pos    = (const float*)d_in[2];
  const float* Wqkv   = (const float*)d_in[3];
  const float* bqkv   = (const float*)d_in[4];
  const float* Wo     = (const float*)d_in[5];
  const float* bo     = (const float*)d_in[6];
  const float* g_attn = (const float*)d_in[7];
  const float* b_attn = (const float*)d_in[8];
  const float* g_ffn  = (const float*)d_in[9];
  const float* b_ffn  = (const float*)d_in[10];
  const float* W1     = (const float*)d_in[11];
  const float* b1     = (const float*)d_in[12];
  const float* W2     = (const float*)d_in[13];
  const float* b2     = (const float*)d_in[14];
  float* out = (float*)d_out;

  char* p = (char*)d_ws;
  auto alloc = [&](size_t bytes) -> void* {
    void* r = (void*)p;
    p += (bytes + 255) & ~(size_t)255;
    return r;
  };
  const size_t ND = (size_t)NROWS * DD;
  float*    z     = (float*)   alloc(ND * 4);
  _Float16* zn    = (_Float16*)alloc(ND * 2);        // reused as hn
  _Float16* qbuf  = (_Float16*)alloc(ND * 2);
  _Float16* kbuf  = (_Float16*)alloc(ND * 2);
  _Float16* vTbuf = (_Float16*)alloc(ND * 2);
  _Float16* attnh = (_Float16*)alloc(ND * 2);
  _Float16* h1    = (_Float16*)alloc(ND * 4 * 2);    // [NROWS, 4D] f16
  _Float16* wq_h  = (_Float16*)alloc((size_t)3 * DD * DD * 2);
  _Float16* wo_h  = (_Float16*)alloc((size_t)DD * DD * 2);
  _Float16* w1_h  = (_Float16*)alloc((size_t)4 * DD * DD * 2);
  _Float16* w2_h  = (_Float16*)alloc((size_t)4 * DD * DD * 2);

  // 1) weights -> f16
  cvt_f16_kernel<<<1024, 256, 0, stream>>>(Wqkv, wq_h, 3 * DD * DD);
  cvt_f16_kernel<<<1024, 256, 0, stream>>>(Wo,   wo_h, DD * DD);
  cvt_f16_kernel<<<1024, 256, 0, stream>>>(W1,   w1_h, 4 * DD * DD);
  cvt_f16_kernel<<<1024, 256, 0, stream>>>(W2,   w2_h, 4 * DD * DD);

  // 2) z = concat(mem,chunk)+pos ; zn = LN(z)
  build_z_ln_kernel<<<NROWS, 256, 0, stream>>>(chunk, mem, pos, g_attn, b_attn, z, zn);

  // 3) QKV GEMM (N=3072), split into q (scaled), k, vT
  gemm_bstage_kernel<<<(NROWS / 128) * (3072 / 64), 256, 0, stream>>>(
      zn, wq_h, bqkv, qbuf, kbuf, vTbuf, nullptr, NROWS, 3 * DD, DD, 1);

  // 4) attention
  const size_t attn_lds = (size_t)16 * SP * 2 + 4 * 256 * 4;  // 78,080 B
  attn_kernel<<<BB * HH * (TT / 16), 256, attn_lds, stream>>>(qbuf, kbuf, vTbuf, attnh);

  // 5) z += attn @ Wo^T + bo
  gemm_bstage_kernel<<<(NROWS / 128) * (1024 / 64), 256, 0, stream>>>(
      attnh, wo_h, bo, nullptr, nullptr, nullptr, z, NROWS, DD, DD, 2);

  // 6) hn = LN(z)
  ln_kernel<<<NROWS, 256, 0, stream>>>(z, g_ffn, b_ffn, zn);

  // 7) h1 = relu(hn @ W1^T + b1)   (N=4096)
  gemm_bstage_kernel<<<(NROWS / 128) * (4096 / 64), 256, 0, stream>>>(
      zn, w1_h, b1, h1, nullptr, nullptr, nullptr, NROWS, 4 * DD, DD, 3);

  // 8) z += h1 @ W2^T + b2   (K=4096)
  gemm_bstage_kernel<<<(NROWS / 128) * (1024 / 64), 256, 0, stream>>>(
      h1, w2_h, b2, nullptr, nullptr, nullptr, z, NROWS, DD, 4 * DD, 2);

  // 9) outputs
  write_out_kernel<<<2048, 256, 0, stream>>>(z, out);
}
